// Attention_7593502179623
// MI455X (gfx1250) — matrix operand order repaired and checked by reading, not compiled, not run on hardware
//
#include <hip/hip_runtime.h>

typedef __attribute__((ext_vector_type(16))) __bf16 v16bf;
typedef __attribute__((ext_vector_type(8)))  __bf16 v8bf;
typedef __attribute__((ext_vector_type(8)))  float  v8f;

#define WMMA_BF16(a, b, c) \
  __builtin_amdgcn_wmma_f32_16x16x32_bf16(false, (a), false, (b), (short)0, (c), false, false)

static constexpr int Bsz  = 8;
static constexpr int Nseq = 2048;
static constexpr int Cdim = 768;
static constexpr int H    = 12;
static constexpr int Dh   = 64;
static constexpr int Mrow = Bsz * Nseq;   // 16384
static constexpr int N2   = 3 * Cdim;     // 2304

// concat two 8-wide bf16 vectors into a 16-wide fragment
static __device__ inline v16bf cat8(v8bf a, v8bf b) {
  return __builtin_shufflevector(a, b, 0, 1, 2, 3, 4, 5, 6, 7,
                                       8, 9, 10, 11, 12, 13, 14, 15);
}
// convert two 8-wide f32 vectors into a 16-wide bf16 fragment
static __device__ inline v16bf cvt2(v8f f0, v8f f1) {
  v16bf r;
#pragma unroll
  for (int e = 0; e < 8; ++e) { r[e] = (__bf16)f0[e]; r[e + 8] = (__bf16)f1[e]; }
  return r;
}

// ---------------------------------------------------------------------------
// Kernel 1: qkv = x @ w_qkv^T  -> Q[B,H,N,D] bf16, K[B,H,N,D] bf16,
//           V^T[B,H,D,N] bf16.  Per-wave 32x32 tile, K chunks of 32.
// ---------------------------------------------------------------------------
__global__ __launch_bounds__(128) void qkv_gemm_kernel(
    const float* __restrict__ x, const float* __restrict__ w,
    __bf16* __restrict__ qb, __bf16* __restrict__ kb, __bf16* __restrict__ vtb) {
  const int lane = threadIdx.x & 31;
  const int wave = (int)((blockIdx.x * blockDim.x + threadIdx.x) >> 5);
  const int lo = lane & 15, hi = lane >> 4;
  const int tiles_n = N2 / 32;                  // 72
  const int tm = wave / tiles_n, tn = wave % tiles_n;
  const int m0 = tm * 32, n0 = tn * 32;

  v8f zero = {0.f, 0.f, 0.f, 0.f, 0.f, 0.f, 0.f, 0.f};
  v8f c00 = zero, c01 = zero, c10 = zero, c11 = zero;

  const float* arow0 = x + (size_t)(m0 + lo) * Cdim + hi * 16;
  const float* arow1 = x + (size_t)(m0 + 16 + lo) * Cdim + hi * 16;
  const float* brow0 = w + (size_t)(n0 + lo) * Cdim + hi * 16;
  const float* brow1 = w + (size_t)(n0 + 16 + lo) * Cdim + hi * 16;

  for (int ks = 0; ks < Cdim; ks += 32) {
    // A: rows of x, two 8-wide f32 chunks per fragment (32B-aligned)
    v16bf a0 = cvt2(*(const v8f*)(arow0 + ks), *(const v8f*)(arow0 + ks + 8));
    v16bf a1 = cvt2(*(const v8f*)(arow1 + ks), *(const v8f*)(arow1 + ks + 8));
    // B: rows of w_qkv, 16 contiguous f32 per lane (64B-aligned)
    v16bf b0 = cvt2(*(const v8f*)(brow0 + ks), *(const v8f*)(brow0 + ks + 8));
    v16bf b1 = cvt2(*(const v8f*)(brow1 + ks), *(const v8f*)(brow1 + ks + 8));
    c00 = WMMA_BF16(a0, b0, c00);
    c01 = WMMA_BF16(a0, b1, c01);
    c10 = WMMA_BF16(a1, b0, c10);
    c11 = WMMA_BF16(a1, b1, c11);
  }

#pragma unroll
  for (int i = 0; i < 2; ++i) {
#pragma unroll
    for (int j = 0; j < 2; ++j) {
      v8f c = (i == 0) ? ((j == 0) ? c00 : c01) : ((j == 0) ? c10 : c11);
      // wave-uniform: a 16-wide subtile never crosses a 768-column boundary
      const int nbase = n0 + j * 16;
      const int which = nbase / Cdim;           // 0=q, 1=k, 2=v (scalar branch)
      const int cbase = nbase - which * Cdim;
      const int h = (cbase + 0) >> 6;           // 16-wide subtile stays in one head? no:
      // head index can vary with lo (64-wide heads, 16-wide subtile -> uniform):
      // cbase is a multiple of 16; head = (cbase+lo)>>6 == cbase>>6 (uniform).
      const int dbase = cbase & 63;             // multiple of 16
#pragma unroll
      for (int v = 0; v < 8; ++v) {
        const int m = m0 + i * 16 + v + hi * 8;
        const float val = c[v];
        const int d = dbase + lo;
        const int bidx = m >> 11;               // / Nseq
        const int n = m & (Nseq - 1);
        const size_t bhh = (size_t)bidx * H + h;
        if (which == 0)      qb[(bhh * Nseq + n) * Dh + d] = (__bf16)val;
        else if (which == 1) kb[(bhh * Nseq + n) * Dh + d] = (__bf16)val;
        else                 vtb[(bhh * Dh + d) * Nseq + n] = (__bf16)val;
      }
    }
  }
}

// ---------------------------------------------------------------------------
// Kernel 2: flash attention. One wave per 16-query tile; online softmax over
// 2048 keys in blocks of 32. 4 WMMAs for S, LDS transpose of P, 4 WMMAs for O.
// ---------------------------------------------------------------------------
__global__ __launch_bounds__(128) void attn_kernel(
    const __bf16* __restrict__ qb, const __bf16* __restrict__ kb,
    const __bf16* __restrict__ vtb, __bf16* __restrict__ aob) {
  __shared__ __align__(32) __bf16 plds[4][16 * 32];

  const int lane = threadIdx.x & 31;
  const int wib  = threadIdx.x >> 5;
  const int gwave = (int)blockIdx.x * 4 + wib;
  const int lo = lane & 15, hi = lane >> 4;
  const int NT = Nseq / 16;                     // 128
  const int bh = gwave / NT, nt = gwave % NT;
  const int b = bh / H, h = bh % H;

  const __bf16* Q  = qb  + ((size_t)bh * Nseq + nt * 16) * Dh;
  const __bf16* Kp = kb  + (size_t)bh * Nseq * Dh;
  const __bf16* Vt = vtb + (size_t)bh * Dh * Nseq;

  // Q fragments: d = 0..31 (q0), d = 32..63 (q1)
  const __bf16* qrow = Q + (size_t)lo * Dh + hi * 16;
  v16bf q0 = cat8(*(const v8bf*)(qrow),      *(const v8bf*)(qrow + 8));
  v16bf q1 = cat8(*(const v8bf*)(qrow + 32), *(const v8bf*)(qrow + 40));

  float m_[8], l_[8];
#pragma unroll
  for (int v = 0; v < 8; ++v) { m_[v] = -3.0e38f; l_[v] = 0.f; }
  v8f zero = {0.f, 0.f, 0.f, 0.f, 0.f, 0.f, 0.f, 0.f};
  v8f o[4] = {zero, zero, zero, zero};

  const float scale = 0.125f;                   // 1/sqrt(64)
  __bf16* pt = &plds[wib][0];
  const __bf16* pr = pt + lo * 32 + hi * 16;

  for (int j0 = 0; j0 < Nseq; j0 += 32) {
    // K fragments: 2 key sub-tiles x 2 d-chunks (all 32B-aligned b128 pairs)
    const __bf16* kr0 = Kp + (size_t)(j0 + lo) * Dh + hi * 16;
    const __bf16* kr1 = Kp + (size_t)(j0 + 16 + lo) * Dh + hi * 16;
    v16bf k00 = *(const v16bf*)(kr0);
    v16bf k01 = *(const v16bf*)(kr0 + 32);
    v16bf k10 = *(const v16bf*)(kr1);
    v16bf k11 = *(const v16bf*)(kr1 + 32);

    v8f s0 = zero, s1 = zero;
    s0 = WMMA_BF16(q0, k00, s0);
    s0 = WMMA_BF16(q1, k01, s0);
    s1 = WMMA_BF16(q0, k10, s1);
    s1 = WMMA_BF16(q1, k11, s1);

    // online softmax update + write P tile (bf16) to this wave's LDS slice
#pragma unroll
    for (int v = 0; v < 8; ++v) {
      float a0 = s0[v] * scale, a1 = s1[v] * scale;
      float rm = fmaxf(a0, a1);
      rm = fmaxf(rm, __shfl_xor(rm, 1, 16));
      rm = fmaxf(rm, __shfl_xor(rm, 2, 16));
      rm = fmaxf(rm, __shfl_xor(rm, 4, 16));
      rm = fmaxf(rm, __shfl_xor(rm, 8, 16));
      float nm = fmaxf(m_[v], rm);
      float corr = __expf(m_[v] - nm);
      float p0 = __expf(a0 - nm);
      float p1 = __expf(a1 - nm);
      float rs = p0 + p1;
      rs += __shfl_xor(rs, 1, 16);
      rs += __shfl_xor(rs, 2, 16);
      rs += __shfl_xor(rs, 4, 16);
      rs += __shfl_xor(rs, 8, 16);
      l_[v] = l_[v] * corr + rs;
      m_[v] = nm;
      o[0][v] *= corr; o[1][v] *= corr; o[2][v] *= corr; o[3][v] *= corr;
      const int row = v + hi * 8;
      pt[row * 32 + lo]      = (__bf16)p0;
      pt[row * 32 + 16 + lo] = (__bf16)p1;
    }

    // per-wave LDS store->load is HW-ordered; wait + clobber pins the compiler
    asm volatile("s_wait_dscnt 0" ::: "memory");

    // P as A fragment (16 rows x 32 keys) from LDS
    v16bf pA = cat8(*(const v8bf*)(pr), *(const v8bf*)(pr + 8));

    // O += P @ V  (V^T layout -> contiguous 32B loads)
#pragma unroll
    for (int db = 0; db < 4; ++db) {
      const __bf16* vr = Vt + (size_t)(db * 16 + lo) * Nseq + j0 + hi * 16;
      v16bf vf = *(const v16bf*)vr;
      o[db] = WMMA_BF16(pA, vf, o[db]);
    }
  }

  // normalize and write attn_out [B, N, C] (heads merged)
#pragma unroll
  for (int v = 0; v < 8; ++v) {
    const float inv = 1.0f / l_[v];
    const int row = v + hi * 8;
    const int n = nt * 16 + row;
    const size_t base = ((size_t)b * Nseq + n) * Cdim + h * Dh;
#pragma unroll
    for (int db = 0; db < 4; ++db) {
      aob[base + db * 16 + lo] = (__bf16)(o[db][v] * inv);
    }
  }
}

// ---------------------------------------------------------------------------
// Kernel 3: out = attn_out @ w_proj^T + b_proj (f32 out)
// ---------------------------------------------------------------------------
__global__ __launch_bounds__(128) void proj_gemm_kernel(
    const __bf16* __restrict__ a, const float* __restrict__ w,
    const float* __restrict__ bias, float* __restrict__ out) {
  const int lane = threadIdx.x & 31;
  const int wave = (int)((blockIdx.x * blockDim.x + threadIdx.x) >> 5);
  const int lo = lane & 15, hi = lane >> 4;
  const int tiles_n = Cdim / 32;                // 24
  const int tm = wave / tiles_n, tn = wave % tiles_n;
  const int m0 = tm * 32, n0 = tn * 32;

  v8f zero = {0.f, 0.f, 0.f, 0.f, 0.f, 0.f, 0.f, 0.f};
  v8f c00 = zero, c01 = zero, c10 = zero, c11 = zero;

  const __bf16* arow0 = a + (size_t)(m0 + lo) * Cdim + hi * 16;
  const __bf16* arow1 = a + (size_t)(m0 + 16 + lo) * Cdim + hi * 16;
  const float*  brow0 = w + (size_t)(n0 + lo) * Cdim + hi * 16;
  const float*  brow1 = w + (size_t)(n0 + 16 + lo) * Cdim + hi * 16;

  for (int ks = 0; ks < Cdim; ks += 32) {
    v16bf a0 = cat8(*(const v8bf*)(arow0 + ks), *(const v8bf*)(arow0 + ks + 8));
    v16bf a1 = cat8(*(const v8bf*)(arow1 + ks), *(const v8bf*)(arow1 + ks + 8));
    v16bf b0 = cvt2(*(const v8f*)(brow0 + ks), *(const v8f*)(brow0 + ks + 8));
    v16bf b1 = cvt2(*(const v8f*)(brow1 + ks), *(const v8f*)(brow1 + ks + 8));
    c00 = WMMA_BF16(a0, b0, c00);
    c01 = WMMA_BF16(a0, b1, c01);
    c10 = WMMA_BF16(a1, b0, c10);
    c11 = WMMA_BF16(a1, b1, c11);
  }

#pragma unroll
  for (int i = 0; i < 2; ++i) {
#pragma unroll
    for (int j = 0; j < 2; ++j) {
      v8f c = (i == 0) ? ((j == 0) ? c00 : c01) : ((j == 0) ? c10 : c11);
      const int n = n0 + j * 16 + lo;
      const float bv = bias[n];
#pragma unroll
      for (int v = 0; v < 8; ++v) {
        const int m = m0 + i * 16 + v + hi * 8;
        out[(size_t)m * Cdim + n] = c[v] + bv;
      }
    }
  }
}

// ---------------------------------------------------------------------------
extern "C" void kernel_launch(void* const* d_in, const int* in_sizes, int n_in,
                              void* d_out, int out_size, void* d_ws, size_t ws_size,
                              hipStream_t stream) {
  const float* x      = (const float*)d_in[0];
  const float* w_qkv  = (const float*)d_in[1];
  const float* w_proj = (const float*)d_in[2];
  const float* b_proj = (const float*)d_in[3];
  float* out = (float*)d_out;

  // workspace partition (bf16): Q, K, V^T, attn_out -> 4 * 24 MB = 96 MB
  const size_t elems = (size_t)Bsz * H * Nseq * Dh;   // 12,582,912
  __bf16* qb  = (__bf16*)d_ws;
  __bf16* kb  = qb + elems;
  __bf16* vtb = kb + elems;
  __bf16* aob = vtb + elems;

  {
    const int waves = (Mrow / 32) * (N2 / 32);        // 36864
    qkv_gemm_kernel<<<waves / 4, 128, 0, stream>>>(x, w_qkv, qb, kb, vtb);
  }
  {
    const int waves = Bsz * H * (Nseq / 16);          // 12288
    attn_kernel<<<waves / 4, 128, 0, stream>>>(qb, kb, vtb, aob);
  }
  {
    const int waves = (Mrow / 32) * (Cdim / 32);      // 12288
    proj_gemm_kernel<<<waves / 4, 128, 0, stream>>>(aob, w_proj, b_proj, out);
  }
}
